// DongTaiBaGuaZhen_2362232013031
// MI455X (gfx1250) — matrix operand order repaired and checked by reading, compiled, not run on hardware
//
#include <hip/hip_runtime.h>
#include <hip/hip_bf16.h>

typedef __attribute__((ext_vector_type(16))) __bf16 bf16x16;
typedef __attribute__((ext_vector_type(8)))  float  v8f;
typedef __attribute__((ext_vector_type(4)))  unsigned int u32x4;

#define B_  4
#define S_  4096
#define D_  1024
#define H_  8
#define P_  32
#define K_  128
#define MROWS (B_ * S_)       // 16384

#define TM 128
#define TN 128
#define KC 32
#define LDST 40               // ushorts per LDS row (80B, 16B-aligned rows, conflict-free)

__device__ __forceinline__ unsigned short f2bf(float f) {
    unsigned int u = __float_as_uint(f);
    unsigned int r = (u + 0x7FFFu + ((u >> 16) & 1u)) >> 16;   // RNE
    return (unsigned short)r;
}

union Frag { bf16x16 bf; u32x4 q[2]; };

// Async-copy a 128x32 bf16 tile (64B/row) from global into a padded LDS tile.
// 512 b128 transfers, 256 threads x 2. Tracked by ASYNCcnt; no VGPR data path.
__device__ __forceinline__ void async_tile(const unsigned short* __restrict__ G, int ldg,
                                           int row0, int kc,
                                           unsigned short* __restrict__ sm, int tid) {
    #pragma unroll
    for (int it = 0; it < 2; ++it) {
        const int q   = it * 256 + tid;
        const int row = q >> 2;          // 4 x 16B per 32-elem row
        const int qt  = q & 3;
        const unsigned short* gp = G + (size_t)(row0 + row) * ldg + kc + qt * 8;
        const unsigned loff = (unsigned)(size_t)(sm + row * LDST + qt * 8);
        asm volatile("global_load_async_to_lds_b128 %0, %1, off"
                     :: "v"(loff), "v"(gp) : "memory");
    }
}

__device__ __forceinline__ void wait_async0() {
    asm volatile("s_wait_asynccnt 0x0" ::: "memory");
}

// C[M,N] = A[M,Kd] @ W[N,Kd]^T  (A, W are bf16 in global), epilogue:
//   MODE 0: C *= aux0[col]                   (precomputed cos modulation)
//   MODE 1: C += aux0[col] + aux1[row*N+col] (bias + residual)
template<int MODE>
__global__ __launch_bounds__(256)
void gemm_bf16_wmma(const unsigned short* __restrict__ A, const unsigned short* __restrict__ W,
                    float* __restrict__ C, int Kd, int N,
                    const float* __restrict__ aux0, const float* __restrict__ aux1) {
    __shared__ __align__(16) unsigned short smA[2][TM * LDST];
    __shared__ __align__(16) unsigned short smB[2][TN * LDST];

    const int tid  = threadIdx.x;
    const int lane = tid & 31;
    const int wave = tid >> 5;                 // 0..7 -> M strip
    const int m0   = blockIdx.y * TM;
    const int n0   = blockIdx.x * TN;

    v8f acc[8];
    #pragma unroll
    for (int t = 0; t < 8; ++t)
        #pragma unroll
        for (int i = 0; i < 8; ++i) acc[t][i] = 0.0f;

    async_tile(A, Kd, m0, 0, smA[0], tid);
    async_tile(W, Kd, n0, 0, smB[0], tid);
    wait_async0();
    __syncthreads();

    const int mrow  = wave * 16 + (lane & 15);
    const int halfA = (lane >> 4) * 16;        // A: K 0-15 lanes 0-15, 16-31 lanes 16-31
    const int halfB = (lane >> 4) * 16;        // B: K 0-15 lanes 0-15, 16-31 lanes 16-31
    const int nlo   = lane & 15;

    int buf = 0;
    for (int kc = 0; kc < Kd; kc += KC, buf ^= 1) {
        const int kn = kc + KC;
        if (kn < Kd) {                         // async-stage next chunk into other buffer
            async_tile(A, Kd, m0, kn, smA[buf ^ 1], tid);
            async_tile(W, Kd, n0, kn, smB[buf ^ 1], tid);
        }
        if (kc + 2 * KC < Kd) {                // prefetch chunk after next into cache
            if (tid < TM) __builtin_prefetch(A + (size_t)(m0 + tid) * Kd + kc + 2 * KC, 0, 0);
            else          __builtin_prefetch(W + (size_t)(n0 + tid - TM) * Kd + kc + 2 * KC, 0, 0);
        }

        const unsigned short* sA = smA[buf];
        const unsigned short* sB = smB[buf];

        Frag a;
        a.q[0] = *(const u32x4*)&sA[mrow * LDST + halfA];
        a.q[1] = *(const u32x4*)&sA[mrow * LDST + halfA + 8];

        // software-pipelined B fragments: load t+1 before issuing wmma t so the
        // ds_load latency overlaps matrix math instead of stalling on dscnt==0
        Frag bcur, bnxt;
        bcur.q[0] = *(const u32x4*)&sB[nlo * LDST + halfB];
        bcur.q[1] = *(const u32x4*)&sB[nlo * LDST + halfB + 8];
        #pragma unroll
        for (int t = 0; t < 8; ++t) {
            if (t < 7) {
                const int nrow = (t + 1) * 16 + nlo;
                bnxt.q[0] = *(const u32x4*)&sB[nrow * LDST + halfB];
                bnxt.q[1] = *(const u32x4*)&sB[nrow * LDST + halfB + 8];
            }
            acc[t] = __builtin_amdgcn_wmma_f32_16x16x32_bf16(
                false, a.bf, false, bcur.bf, (short)0, acc[t], false, false);
            bcur = bnxt;
        }

        wait_async0();                          // own copies done before signaling
        __syncthreads();
    }

    // C layout: row = m0 + wave*16 + v + 8*(lane>=16), col = n0 + t*16 + (lane&15)
    const int mbase = m0 + wave * 16 + (lane >> 4) * 8;
    #pragma unroll
    for (int t = 0; t < 8; ++t) {
        const int col = n0 + t * 16 + nlo;
        const float aval = aux0[col];
        #pragma unroll
        for (int v = 0; v < 8; ++v) {
            const int row = mbase + v;
            float val = acc[t][v];
            if (MODE == 0) val *= aval;
            else           val += aval + aux1[(size_t)row * N + col];
            C[(size_t)row * N + col] = val;
        }
    }
}

// One float4 -> 4 bf16 per thread
__global__ __launch_bounds__(256)
void f32_to_bf16_kernel(const float* __restrict__ src, unsigned short* __restrict__ dst, int n4) {
    const int i = blockIdx.x * 256 + threadIdx.x;
    if (i < n4) {
        float4 v = ((const float4*)src)[i];
        uint2 p;
        p.x = (unsigned)f2bf(v.x) | ((unsigned)f2bf(v.y) << 16);
        p.y = (unsigned)f2bf(v.z) | ((unsigned)f2bf(v.w) << 16);
        ((uint2*)dst)[i] = p;
    }
}

__global__ __launch_bounds__(256)
void cosmod_kernel(const float* __restrict__ freqs, float* __restrict__ cm) {
    const int i = blockIdx.x * 256 + threadIdx.x;
    if (i < D_) cm[i] = cosf(freqs[i] * 3.14159265358979323846f);
}

// summary[b, c] = mean_s heads[b*S + s, c]  via partial atomic adds
__global__ __launch_bounds__(256)
void summary_kernel(const float* __restrict__ Hm, float* __restrict__ summary) {
    const int b  = blockIdx.x;
    const int c  = blockIdx.y * 256 + threadIdx.x;
    const int s0 = blockIdx.z * 256;
    const float* p = Hm + ((size_t)(b * S_ + s0)) * D_ + c;
    float s = 0.0f;
    for (int i = 0; i < 256; ++i) s += p[(size_t)i * D_];
    atomicAdd(&summary[b * D_ + c], s * (1.0f / (float)S_));
}

__global__ __launch_bounds__(256)
void impedance_kernel(const float* __restrict__ summary,
                      const float* __restrict__ pol_W, const float* __restrict__ pol_b,
                      const float* __restrict__ W1, const float* __restrict__ b1,
                      const float* __restrict__ W2, const float* __restrict__ b2,
                      float* __restrict__ imp_out, float* __restrict__ coef) {
    __shared__ float pol[B_ * H_ * P_];      // 1024
    __shared__ float rnorm[B_ * H_];         // 32
    const int tid = threadIdx.x;

    for (int idx = tid; idx < B_ * H_ * P_; idx += 256) {
        const int b = idx >> 8, h = (idx >> 5) & 7, p = idx & 31;
        const float* s = summary + b * D_ + h * K_;
        const float* w = pol_W + (size_t)(h * P_ + p) * K_;
        float d = 0.0f;
        for (int k = 0; k < K_; ++k) d += s[k] * w[k];
        pol[idx] = tanhf(d + pol_b[h * P_ + p]);
    }
    __syncthreads();
    for (int bh = tid; bh < B_ * H_; bh += 256) {
        float ss = 0.0f;
        for (int p = 0; p < P_; ++p) { float v = pol[bh * P_ + p]; ss += v * v; }
        rnorm[bh] = 1.0f / fmaxf(sqrtf(ss), 1e-12f);
    }
    __syncthreads();
    for (int idx = tid; idx < B_ * H_ * P_; idx += 256) pol[idx] *= rnorm[idx >> 5];
    __syncthreads();

    if (tid < B_ * H_ * H_) {                // 256 entries
        const int e = tid;
        const int b = e >> 6, i = (e >> 3) & 7, j = e & 7;
        const float* pi_ = &pol[b * 256 + i * 32];
        const float* pj  = &pol[b * 256 + j * 32];
        float d = 0.0f;
        for (int p = 0; p < P_; ++p) d += pi_[p] * pj[p];
        float accv = 0.0f;
        for (int c = 0; c < 16; ++c) {
            float z = d * W1[c] + b1[c];
            float g = 0.5f * z * (1.0f + erff(z * 0.7071067811865476f)); // exact gelu
            accv += g * W2[c];
        }
        accv += b2[0];
        float sp = (accv > 20.0f) ? accv : log1pf(expf(accv));
        float imp = (i == j) ? 0.0f : sp;
        imp_out[e] = imp;
        coef[e]    = (i == j) ? 0.0f : 0.1f / (1.0f + imp);
    }
}

// merged_bf16[bs, i*K+k] = bf16( h_i + scale(s) * sum_j coef[b,i,j]*h_j )
__global__ __launch_bounds__(256)
void mix_kernel(const float* __restrict__ Hm, unsigned short* __restrict__ Mb,
                const float* __restrict__ coef, const int* __restrict__ causal) {
    __shared__ float cf[H_ * H_];
    const int bs = blockIdx.x * 2 + (threadIdx.x >> 7);
    const int k  = threadIdx.x & 127;
    const int b  = bs >> 12;
    const int s  = bs & (S_ - 1);
    if (threadIdx.x < H_ * H_) cf[threadIdx.x] = coef[b * H_ * H_ + threadIdx.x];
    __syncthreads();

    const float* rowp = Hm + (size_t)bs * D_ + k;
    unsigned short* outp = Mb + (size_t)bs * D_ + k;
    float h[H_];
    #pragma unroll
    for (int j = 0; j < H_; ++j) h[j] = rowp[j * K_];
    const float scale = (*causal != 0) ? (float)(s + 1) * (1.0f / (float)S_) : 1.0f;
    #pragma unroll
    for (int i = 0; i < H_; ++i) {
        float t = 0.0f;
        #pragma unroll
        for (int j = 0; j < H_; ++j) t += cf[i * H_ + j] * h[j];
        outp[i * K_] = f2bf(h[i] + scale * t);
    }
}

__global__ __launch_bounds__(256)
void layernorm_kernel(const float* __restrict__ Y, const float* __restrict__ g,
                      const float* __restrict__ bta, float* __restrict__ out) {
    __shared__ float red[256], red2[256];
    const int r = blockIdx.x;
    const int c = threadIdx.x * 4;
    const float* y = Y + (size_t)r * D_;
    float4 vv = *(const float4*)(y + c);
    red[threadIdx.x]  = vv.x + vv.y + vv.z + vv.w;
    red2[threadIdx.x] = vv.x * vv.x + vv.y * vv.y + vv.z * vv.z + vv.w * vv.w;
    __syncthreads();
    for (int off = 128; off > 0; off >>= 1) {
        if (threadIdx.x < off) {
            red[threadIdx.x]  += red[threadIdx.x + off];
            red2[threadIdx.x] += red2[threadIdx.x + off];
        }
        __syncthreads();
    }
    const float mu  = red[0] * (1.0f / (float)D_);
    const float var = red2[0] * (1.0f / (float)D_) - mu * mu;
    const float inv = rsqrtf(var + 1e-5f);
    float4 go = *(const float4*)(g + c);
    float4 bo = *(const float4*)(bta + c);
    float4 o;
    o.x = (vv.x - mu) * inv * go.x + bo.x;
    o.y = (vv.y - mu) * inv * go.y + bo.y;
    o.z = (vv.z - mu) * inv * go.z + bo.z;
    o.w = (vv.w - mu) * inv * go.w + bo.w;
    *(float4*)(out + (size_t)r * D_ + c) = o;
}

extern "C" void kernel_launch(void* const* d_in, const int* in_sizes, int n_in,
                              void* d_out, int out_size, void* d_ws, size_t ws_size,
                              hipStream_t stream) {
    (void)in_sizes; (void)n_in; (void)out_size; (void)ws_size;
    const float* x       = (const float*)d_in[0];
    const float* W_proj  = (const float*)d_in[1];
    const float* freqs   = (const float*)d_in[2];
    const float* pol_W   = (const float*)d_in[3];
    const float* pol_b   = (const float*)d_in[4];
    const float* iW1     = (const float*)d_in[5];
    const float* ib1     = (const float*)d_in[6];
    const float* iW2     = (const float*)d_in[7];
    const float* ib2     = (const float*)d_in[8];
    const float* out_W   = (const float*)d_in[9];
    const float* out_b   = (const float*)d_in[10];
    const float* ln_g    = (const float*)d_in[11];
    const float* ln_b    = (const float*)d_in[12];
    const int*   causal  = (const int*)d_in[13];

    float* out     = (float*)d_out;
    float* imp_out = out + (size_t)MROWS * D_;            // normed then impedance

    // Workspace layout (aliased across phases):
    //   [0,   64MB): Hbuf (f32 heads)        -> reused as Ybuf after mix
    //   [64,  96MB): xbf  (bf16 x)           -> reused as Mbf (bf16 merged)
    //   [96,  98MB): Wpbf (bf16 W_proj)
    //   [98, 100MB): oWbf (bf16 out_W)
    //   [100MB,...): summary / coef / cosmod
    char* ws = (char*)d_ws;
    float*          Hbuf    = (float*)ws;
    float*          Ybuf    = (float*)ws;                               // alias (Hbuf dead)
    unsigned short* xbf     = (unsigned short*)(ws + (size_t)67108864);
    unsigned short* Mbf     = xbf;                                      // alias (xbf dead)
    unsigned short* Wpbf    = (unsigned short*)(ws + (size_t)100663296);
    unsigned short* oWbf    = (unsigned short*)(ws + (size_t)102760448);
    float*          summary = (float*)(ws + (size_t)104857600);
    float*          coef    = (float*)(ws + (size_t)104857600 + 16384);
    float*          cosmod  = (float*)(ws + (size_t)104857600 + 16384 + 1024);

    dim3 blk(256);
    dim3 grid(D_ / TN, MROWS / TM);   // (8, 128)

    f32_to_bf16_kernel<<<(MROWS * D_ / 4) / 256, blk, 0, stream>>>(x, xbf, MROWS * D_ / 4);
    f32_to_bf16_kernel<<<(D_ * D_ / 4) / 256, blk, 0, stream>>>(W_proj, Wpbf, D_ * D_ / 4);
    f32_to_bf16_kernel<<<(D_ * D_ / 4) / 256, blk, 0, stream>>>(out_W, oWbf, D_ * D_ / 4);
    cosmod_kernel<<<D_ / 256, blk, 0, stream>>>(freqs, cosmod);

    gemm_bf16_wmma<0><<<grid, blk, 0, stream>>>(xbf, Wpbf, Hbuf, D_, D_, cosmod, nullptr);
    hipMemsetAsync(summary, 0, B_ * D_ * sizeof(float), stream);
    summary_kernel<<<dim3(B_, D_ / 256, S_ / 256), blk, 0, stream>>>(Hbuf, summary);
    impedance_kernel<<<1, blk, 0, stream>>>(summary, pol_W, pol_b, iW1, ib1, iW2, ib2,
                                            imp_out, coef);
    mix_kernel<<<MROWS / 2, blk, 0, stream>>>(Hbuf, Mbf, coef, causal);
    gemm_bf16_wmma<1><<<grid, blk, 0, stream>>>(Mbf, oWbf, Ybuf, D_, D_, out_b, x);
    layernorm_kernel<<<MROWS, blk, 0, stream>>>(Ybuf, ln_g, ln_b, out);
}
